// PointwiseAggregatedAttention_46763603919365
// MI455X (gfx1250) — matrix operand order repaired and checked by reading, not compiled, not run on hardware
//
#include <hip/hip_runtime.h>
#include <hip/hip_bf16.h>

typedef __attribute__((ext_vector_type(16))) _Float16 v16h;
typedef __attribute__((ext_vector_type(8)))  _Float16 v8h;
typedef __attribute__((ext_vector_type(4)))  _Float16 v4h;
typedef __attribute__((ext_vector_type(2)))  _Float16 h2;
typedef __attribute__((ext_vector_type(8)))  float    v8f;

#define NUM_BUCKETS 32
#define NUM_HEADS   16
#define D_MODEL     1024
#define HEAD_DIM    64
#define SEQ         2048
#define BATCH       4
#define QT          128    // queries per workgroup (8 waves x 16)
#define KT          64     // keys per LDS stage
#define LDSW        72     // padded LDS row stride in halves (64 + 8)
#define BIASN       4096   // reversed bias table: biasr[j] = bias(max(2048-j,0))

#if __has_builtin(__builtin_amdgcn_tanhf)
  #define FAST_TANH(x) __builtin_amdgcn_tanhf(x)
  #define HAVE_FAST_TANH 1
#elif __has_builtin(__builtin_amdgcn_tanh_f32)
  #define FAST_TANH(x) __builtin_amdgcn_tanh_f32(x)
  #define HAVE_FAST_TANH 1
#else
  #define HAVE_FAST_TANH 0
#endif

template <typename T, typename F>
__device__ inline T bcast(F x) { return __builtin_bit_cast(T, x); }

__device__ inline float silu_f(float x) {
#if HAVE_FAST_TANH
    // x*sigmoid(x) = t + t*tanh(t), t = x/2  (v_mul + v_tanh + v_fma)
    const float t = 0.5f * x;
    return __builtin_fmaf(t, FAST_TANH(t), t);
#else
    return x * __builtin_amdgcn_rcpf(1.0f + __expf(-x));
#endif
}

// O = silu(Q K^T + bias) V per (batch, head).
// GEMM1 computes S^T = K Q^T + bias (bias folded into the WMMA C operand,
// table stored reversed so in-tile loads are ascending/contiguous and land
// directly in the C register tuple); the f32 D-fragment (lane = query,
// VGPR = key) packs in-lane via v_cvt_pk_rtz_f16_f32 into the A-fragment of
// GEMM2 O += P V. All 8 GEMM1 WMMAs of a block issue before any silu so the
// scheduler fills WMMA hazard slots with fragment loads.
__global__ __launch_bounds__(256)
void paa_kernel(const float* __restrict__ vin,
                const float* __restrict__ kin,
                const float* __restrict__ qin,
                const float* __restrict__ rab,
                float* __restrict__ out)
{
    __shared__ float    biasr[BIASN];           // reversed distance -> bias
    __shared__ _Float16 Kls[KT * LDSW];         // [key][dim] f16
    __shared__ _Float16 Vls[HEAD_DIM * LDSW];   // [dim][key] f16 (transposed)

    const int tid  = threadIdx.x;
    const int lane = tid & 31;
    const int wave = tid >> 5;
    const int lrow = lane & 15;
    const int lhi  = lane >> 4;

    const int qblk = blockIdx.x;     // 0..15
    const int h    = blockIdx.y;     // 0..15
    const int b    = blockIdx.z;     // 0..3
    const int hbase = h * HEAD_DIM;

    // ---- reversed, clamp-free T5 bias table ----
    for (int j = tid; j < BIASN; j += 256) {
        int d = 2048 - j; if (d < 0) d = 0;
        int bucket;
        if (d < 16) {
            bucket = d;
        } else {
            int lb = 16 + (int)(__logf((float)d * (1.0f / 16.0f)) *
                                (16.0f / __logf(8.0f)));
            bucket = lb < (NUM_BUCKETS - 1) ? lb : (NUM_BUCKETS - 1);
        }
        biasr[j] = rab[bucket * NUM_HEADS + h];
    }

    // ---- Q fragments (B-matrix layout), resident in registers ----
    const int qrow = qblk * QT + wave * 16 + lrow;
    const float* qp = qin + ((size_t)(b * SEQ + qrow) * D_MODEL + hbase);
    v16h qfrag[2];
    #pragma unroll
    for (int c = 0; c < 2; ++c) {
        const float* p = qp + c * 32 + lhi * 8;
        union { v16h v; h2 p[8]; } u;
        #pragma unroll
        for (int j = 0; j < 8; ++j)
            u.p[j] = bcast<h2>(__builtin_amdgcn_cvt_pkrtz(p[2 * j + (j & 4) * 2], p[2 * j + 1 + (j & 4) * 2]));
        qfrag[c] = u.v;
    }

    // ---- staging register tiles (software pipeline), persistent pointers ----
    const int krow  = tid >> 2;          // K: one key row, 16 dims
    const int kquad = (tid & 3) * 16;
    const int vdim  = (tid & 15) * 4;    // V: 4 dims x 4 keys micro-tile
    const int vkey  = (tid >> 4) * 4;

    const float* kp = kin + ((size_t)(b * SEQ + krow) * D_MODEL + hbase + kquad);
    const float* vp = vin + ((size_t)(b * SEQ + vkey) * D_MODEL + hbase + vdim);

    float kreg[16];
    float vreg[4][4];
    auto stage_load = [&]() {
        #pragma unroll
        for (int i = 0; i < 16; ++i) kreg[i] = kp[i];
        #pragma unroll
        for (int j = 0; j < 4; ++j) {
            #pragma unroll
            for (int i = 0; i < 4; ++i) vreg[j][i] = vp[j * D_MODEL + i];
        }
    };

    v8f Ofrag[4] = {};   // 16 queries x 64 dims accumulator

    // tile element i at key base kbase reads biasr[2048 - qrow + kbase + i]
    const int bias0 = 2048 - qrow + lhi * 8;

    stage_load();
    for (int kb = 0; kb < SEQ; kb += KT) {
        __syncthreads();                 // previous block's readers done
        // ---- registers -> f16 LDS (pack pairs, wide DS stores) ----
        {
            union { v8h half8[2]; h2 p[8]; } ku;
            #pragma unroll
            for (int j = 0; j < 8; ++j)
                ku.p[j] = bcast<h2>(__builtin_amdgcn_cvt_pkrtz(kreg[2 * j], kreg[2 * j + 1]));
            _Float16* kd = &Kls[krow * LDSW + kquad];
            *(v8h*)(kd)     = ku.half8[0];
            *(v8h*)(kd + 8) = ku.half8[1];
            #pragma unroll
            for (int i = 0; i < 4; ++i) {          // transposed: keys contiguous
                union { v4h v; h2 p[2]; } vu;
                vu.p[0] = bcast<h2>(__builtin_amdgcn_cvt_pkrtz(vreg[0][i], vreg[1][i]));
                vu.p[1] = bcast<h2>(__builtin_amdgcn_cvt_pkrtz(vreg[2][i], vreg[3][i]));
                *(v4h*)&Vls[(vdim + i) * LDSW + vkey] = vu.v;
            }
        }
        __syncthreads();
        kp += KT * D_MODEL; vp += KT * D_MODEL;
        if (kb + KT < SEQ) stage_load();          // overlap HBM with compute

        // freeze the bias base index in a VGPR so the +2048 table-center and
        // per-block part stay in the base; only tt*16 + j*2 dword constants
        // remain for the ds_load_2addr offset fields.
        int bidx = bias0 + kb;
        asm volatile("" : "+v"(bidx));
        const float* bp_blk = &biasr[bidx];

        // ---- GEMM1: all four 16-key S^T tiles of this 64-key block ----
        v8f st[4];
        #pragma unroll
        for (int tt = 0; tt < 4; ++tt) {
            const int k0 = tt * 16;
            const float* bp = bp_blk + k0;        // ascending, contiguous
            #pragma unroll
            for (int i = 0; i < 8; ++i) st[tt][i] = bp[i];
            const _Float16* kr = &Kls[(k0 + lrow) * LDSW + lhi * 8];
            v16h ka0, ka1;
            #pragma unroll
            for (int i = 0; i < 8; ++i) {
                ka0[i]     = kr[i];
                ka0[8 + i] = kr[16 + i];
                ka1[i]     = kr[32 + i];
                ka1[8 + i] = kr[48 + i];
            }
            st[tt] = __builtin_amdgcn_wmma_f32_16x16x32_f16(
                    false, ka0, false, qfrag[0], (short)0, st[tt], false, false);
            st[tt] = __builtin_amdgcn_wmma_f32_16x16x32_f16(
                    false, ka1, false, qfrag[1], (short)0, st[tt], false, false);
        }

        // ---- silu + pack into the two P A-fragments ----
        union { v16h v[2]; h2 p[16]; } pa;
        #pragma unroll
        for (int tt = 0; tt < 4; ++tt) {
            float s[8];
            #pragma unroll
            for (int i = 0; i < 8; ++i) s[i] = silu_f(st[tt][i]);
            #pragma unroll
            for (int j = 0; j < 4; ++j)
                pa.p[tt * 4 + j] =
                    bcast<h2>(__builtin_amdgcn_cvt_pkrtz(s[2 * j], s[2 * j + 1]));
        }

        // ---- GEMM2: Ofrag[n] += P(16x32) * V(32x16), 2 chunks x 4 tiles ----
        #pragma unroll
        for (int cc = 0; cc < 2; ++cc) {
            #pragma unroll
            for (int n = 0; n < 4; ++n) {
                const _Float16* vr = &Vls[(n * 16 + lrow) * LDSW + cc * 32 + lhi * 8];
                v16h vb;
                #pragma unroll
                for (int i = 0; i < 16; ++i) vb[i] = vr[i + (i & 8)];
                Ofrag[n] = __builtin_amdgcn_wmma_f32_16x16x32_f16(
                        false, pa.v[cc], false, vb, (short)0, Ofrag[n], false, false);
            }
        }
    }

    // ---- store O: lane = dim column (16 consecutive cols per half-wave) ----
    #pragma unroll
    for (int n = 0; n < 4; ++n) {
        const int col = hbase + n * 16 + lrow;
        #pragma unroll
        for (int r = 0; r < 8; ++r) {
            const int qr = qblk * QT + wave * 16 + lhi * 8 + r;
            out[(size_t)(b * SEQ + qr) * D_MODEL + col] = Ofrag[n][r];
        }
    }
}

extern "C" void kernel_launch(void* const* d_in, const int* in_sizes, int n_in,
                              void* d_out, int out_size, void* d_ws, size_t ws_size,
                              hipStream_t stream) {
    const float* v   = (const float*)d_in[0];
    const float* k   = (const float*)d_in[1];
    const float* q   = (const float*)d_in[2];
    const float* rab = (const float*)d_in[3];
    float* out = (float*)d_out;

    dim3 grid(SEQ / QT, NUM_HEADS, BATCH);   // 16 x 16 x 4 = 1024 workgroups
    paa_kernel<<<grid, 256, 0, stream>>>(v, k, q, rab, out);
}
